// QuantumLSTM_65481071401457
// MI455X (gfx1250) — matrix operand order repaired and checked by reading, not compiled, not run on hardware
//
#include <hip/hip_runtime.h>
#include <hip/hip_bf16.h>

typedef __bf16 bf16;
typedef __attribute__((ext_vector_type(16))) __bf16 v16bf;
typedef __attribute__((ext_vector_type(8)))  __bf16 v8bf;
typedef __attribute__((ext_vector_type(4)))  __bf16 v4bf;
typedef __attribute__((ext_vector_type(8)))  float  v8f;

typedef __attribute__((ext_vector_type(4))) unsigned int u32x4;
typedef __attribute__((ext_vector_type(8))) int          i32x8;
typedef __attribute__((ext_vector_type(4))) int          i32x4;

#define SEQ    512
#define BATCH  256
#define IN_DIM 512
#define HID    256
#define CDIM   (IN_DIM + HID)   // 768
#define NGATE  (4 * HID)        // 1024

// ---------------------------------------------------------------------------
// Kernel 0: pack weights to bf16 [1024 x 768] (gate-major rows f,i,u,o),
// fuse bias+rotation into one f32 vector, zero h (bf16) and c (f32) state.
// ---------------------------------------------------------------------------
__global__ void pack_kernel(const float* __restrict__ Wf, const float* __restrict__ bf_, const float* __restrict__ rf,
                            const float* __restrict__ Wi, const float* __restrict__ bi_, const float* __restrict__ ri,
                            const float* __restrict__ Wu, const float* __restrict__ bu_, const float* __restrict__ ru,
                            const float* __restrict__ Wo, const float* __restrict__ bo_, const float* __restrict__ ro,
                            bf16* __restrict__ Wall, float* __restrict__ bias,
                            bf16* __restrict__ hb, float* __restrict__ cst)
{
    int idx = blockIdx.x * blockDim.x + threadIdx.x;
    const float* Ws[4] = {Wf, Wi, Wu, Wo};
    const float* bs[4] = {bf_, bi_, bu_, bo_};
    const float* rs[4] = {rf, ri, ru, ro};
    if (idx < NGATE * CDIM) {
        int row = idx / CDIM, col = idx - row * CDIM;
        int g = row >> 8, r = row & 255;
        Wall[idx] = (bf16)Ws[g][r * CDIM + col];
    }
    if (idx < NGATE) {
        int g = idx >> 8, r = idx & 255;
        bias[idx] = bs[g][r] + rs[g][r];
    }
    if (idx < BATCH * HID) {
        hb[idx]  = (bf16)0.0f;
        cst[idx] = 0.0f;
    }
}

// ---------------------------------------------------------------------------
// Kernel 1: convert fp32 inputs -> bf16 (vectorized 4 at a time).
// ---------------------------------------------------------------------------
__global__ void cvt_x_kernel(const float* __restrict__ X, bf16* __restrict__ Xb)
{
    size_t i = ((size_t)blockIdx.x * blockDim.x + threadIdx.x) * 4;
    float4 v = *(const float4*)(X + i);
    v4bf o;
    o.x = (bf16)v.x; o.y = (bf16)v.y; o.z = (bf16)v.z; o.w = (bf16)v.w;
    *(v4bf*)(Xb + i) = o;
}

// ---------------------------------------------------------------------------
// TDM: DMA a 2D bf16 tile [rows x BK] from global into LDS, with hardware
// row padding so the LDS layout is [rows][LDS_STRIDE] bf16.
// D# layout per CDNA5 ISA.  This toolchain exposes the
// 6-arg builtin: (u32x4 g0, i32x8 g1, i32x4 g2, i32x4 g3, i32x8 g4, i32 cpol).
// ---------------------------------------------------------------------------
#define BM 128
#define BN 128
#define BK 64
#define LDS_STRIDE 72   // 64 data + 8 pad bf16 (TDM pads 4 DWORDs per 32-DWORD row)

__device__ __forceinline__ unsigned lds_addr_of(const void* p)
{
    // Flat LDS address: low 32 bits are the logical LDS address (ISA §10.2).
    return (unsigned)(unsigned long long)p;
}

__device__ __forceinline__ void tdm_load_tile_bf16(
    const bf16* gptr, unsigned lds_off, unsigned rows, unsigned stride_elems)
{
    unsigned long long ga = (unsigned long long)(size_t)gptr;

    u32x4 g0;
    g0.x = 1u;                                   // count=1, is_restore=0, gather off
    g0.y = lds_off;                              // lds_addr (bytes)
    g0.z = (unsigned)(ga & 0xFFFFFFFFu);         // global_addr[31:0]
    g0.w = (unsigned)((ga >> 32) & 0x01FFFFFFu)  // global_addr[56:32]
         | (2u << 30);                           // type = 2 ("image")

    const unsigned td0 = stride_elems;           // tensor_dim0 (elements)
    const unsigned td1 = rows;                   // tensor_dim1
    i32x8 g1;
    g1[0] = (1 << 16)                            // data_size = 1 (2 bytes)
          | (1 << 20)                            // pad_enable
          | (4 << 22)                            // pad_interval: 32 DWORDs (=64 bf16)
          | (3 << 25);                           // pad_amount: 4 DWORDs (=8 bf16)
    g1[1] = (int)((td0 & 0xFFFFu) << 16);        // [47:32] atomic_barrier_addr=0, [63:48]=td0 lo
    g1[2] = (int)((td0 >> 16) | ((td1 & 0xFFFFu) << 16));
    g1[3] = (int)((td1 >> 16) | ((unsigned)BK << 16));   // tile_dim0 = BK
    g1[4] = (int)rows;                           // tile_dim1 = rows, tile_dim2 = 0 (2D)
    g1[5] = (int)(stride_elems);                 // tensor_dim0_stride[31:0]
    g1[6] = 0;                                   // stride0[47:32]=0, stride1[15:0]=0
    g1[7] = 0;                                   // tensor_dim1_stride (unused, 2D)

    i32x4 g2 = (i32x4){0, 0, 0, 0};              // 2D: dims 2/3 unused
    i32x4 g3 = (i32x4){0, 0, 0, 0};
    i32x8 g4 = (i32x8){0, 0, 0, 0, 0, 0, 0, 0};  // extra group (unused for 2D tile)

    __builtin_amdgcn_tensor_load_to_lds(g0, g1, g2, g3, g4, /*cpol=*/0);
}

// ---------------------------------------------------------------------------
// WMMA GEMM:  C[m,n] = sum_k A[m,k] * W[n,k]  (+ addP[m,n] + bias[n])
//   A: bf16 row-major [M x lda], W: bf16 row-major [N x ldw] (B = W^T)
//   Block tile 128x128, BK=64, 8 wave32s (2 x 4), wave tile 64x32 (4x2 WMMAs)
//   Double-buffered LDS staging via Tensor Data Mover (TENSORcnt-tracked).
// ---------------------------------------------------------------------------
__global__ __launch_bounds__(256)
void gemm_wmma_kernel(const bf16* __restrict__ A, int lda,
                      const bf16* __restrict__ W, int ldw,
                      float* __restrict__ C, int ldc,
                      const float* __restrict__ addP,   // optional [M x ldc]
                      const float* __restrict__ bias,   // optional [N]
                      int K)
{
    __shared__ bf16 sA[2][BM][LDS_STRIDE];
    __shared__ bf16 sW[2][BN][LDS_STRIDE];

    const int tid  = threadIdx.x;
    const int lane = tid & 31;
    const int wave = tid >> 5;
    const int wm   = (wave >> 2) * 64;   // 0 / 64
    const int wn   = (wave & 3) * 32;    // 0 / 32 / 64 / 96
    const int l15  = lane & 15;
    const bool hiHalf = (lane & 16) != 0;
    const bool issuer = (wave == 0);     // one wave drives the TDM

    const int m0 = blockIdx.y * BM;
    const int n0 = blockIdx.x * BN;

    v8f acc[4][2];
#pragma unroll
    for (int i = 0; i < 4; ++i)
#pragma unroll
        for (int j = 0; j < 2; ++j)
            acc[i][j] = (v8f){0.f, 0.f, 0.f, 0.f, 0.f, 0.f, 0.f, 0.f};

    const int kT = K / BK;

    if (issuer) {
        tdm_load_tile_bf16(A + (size_t)m0 * lda, lds_addr_of(&sA[0][0][0]), BM, lda);
        tdm_load_tile_bf16(W + (size_t)n0 * ldw, lds_addr_of(&sW[0][0][0]), BN, ldw);
    }

    for (int kt = 0; kt < kT; ++kt) {
        const int buf = kt & 1;
        const bool more = (kt + 1 < kT);
        if (issuer) {
            if (more) {
                tdm_load_tile_bf16(A + (size_t)m0 * lda + (kt + 1) * BK,
                                   lds_addr_of(&sA[buf ^ 1][0][0]), BM, lda);
                tdm_load_tile_bf16(W + (size_t)n0 * ldw + (kt + 1) * BK,
                                   lds_addr_of(&sW[buf ^ 1][0][0]), BN, ldw);
                __builtin_amdgcn_s_wait_tensorcnt(2);  // oldest A+W pair (this buf) done
            } else {
                __builtin_amdgcn_s_wait_tensorcnt(0);
            }
        }
        __syncthreads();   // publish TDM-written LDS tiles to all waves

#pragma unroll
        for (int kk = 0; kk < BK; kk += 32) {
            // A fragments: 16x32 bf16; lanes 0-15 row m, K chunks {0..7,16..23};
            // lanes 16-31 same rows, K chunks {8..15,24..31}.
            const int ak = kk + (hiHalf ? 8 : 0);
            v16bf afr[4];
#pragma unroll
            for (int i = 0; i < 4; ++i) {
                const bf16* p = &sA[buf][wm + i * 16 + l15][ak];
                v8bf lo = *(const v8bf*)p;
                v8bf hi = *(const v8bf*)(p + 16);
                afr[i] = __builtin_shufflevector(lo, hi, 0,1,2,3,4,5,6,7,8,9,10,11,12,13,14,15);
            }
            // B fragments: 32x16 bf16; lane n (0-15) holds K chunks {0..7,16..23} of column n,
            // lanes 16-31 hold K chunks {8..15,24..31}.  Column n of B == row n of W (contiguous).
            const int bk = kk + (hiHalf ? 8 : 0);
            v16bf bfr[2];
#pragma unroll
            for (int j = 0; j < 2; ++j) {
                const bf16* p = &sW[buf][wn + j * 16 + l15][bk];
                v8bf lo = *(const v8bf*)p;
                v8bf hi = *(const v8bf*)(p + 16);
                bfr[j] = __builtin_shufflevector(lo, hi, 0,1,2,3,4,5,6,7,8,9,10,11,12,13,14,15);
            }
#pragma unroll
            for (int i = 0; i < 4; ++i)
#pragma unroll
                for (int j = 0; j < 2; ++j)
                    acc[i][j] = __builtin_amdgcn_wmma_f32_16x16x32_bf16(
                        /*neg_a=*/false, afr[i], /*neg_b=*/false, bfr[j],
                        /*c_mod=*/(short)0, acc[i][j],
                        /*reuse_a=*/false, /*reuse_b=*/false);
        }
        __syncthreads();   // all reads of `buf` done before TDM re-targets it
    }

    // Epilogue: C/D layout — VGPR v: lanes 0-15 -> (M = v, N = lane),
    // lanes 16-31 -> (M = 8+v, N = lane-16).
#pragma unroll
    for (int i = 0; i < 4; ++i) {
#pragma unroll
        for (int j = 0; j < 2; ++j) {
            const int mBase = m0 + wm + i * 16 + (hiHalf ? 8 : 0);
            const int nCol  = n0 + wn + j * 16 + l15;
            const float bv  = bias ? bias[nCol] : 0.0f;
#pragma unroll
            for (int v = 0; v < 8; ++v) {
                const size_t off = (size_t)(mBase + v) * ldc + nCol;
                float val = acc[i][j][v] + bv;
                if (addP) val += addP[off];
                C[off] = val;
            }
        }
    }
}

// ---------------------------------------------------------------------------
// Kernel 3: fused LSTM cell update for one time step.
//   pre[b, 1024] = [f | i | g | o] pre-activations (all sigmoid gates)
// ---------------------------------------------------------------------------
__device__ __forceinline__ float sigmoidf_fast(float x)
{
    return 1.0f / (1.0f + __expf(-x));
}

__global__ void step_update_kernel(const float* __restrict__ pre,
                                   float* __restrict__ c, bf16* __restrict__ hb,
                                   float* __restrict__ out_t,
                                   float* __restrict__ hx_out, float* __restrict__ cx_out,
                                   int last)
{
    const int b = blockIdx.x;
    const int h = threadIdx.x;
    const int i0 = b * NGATE + h;
    const float f  = sigmoidf_fast(pre[i0]);
    const float ig = sigmoidf_fast(pre[i0 + HID]);
    const float g  = sigmoidf_fast(pre[i0 + 2 * HID]);
    const float o  = sigmoidf_fast(pre[i0 + 3 * HID]);
    const int ch = b * HID + h;
    const float cn = f * c[ch] + ig * g;
    const float hn = o * tanhf(cn);
    c[ch]   = cn;
    hb[ch]  = (bf16)hn;
    out_t[ch] = hn;
    if (last) {
        hx_out[ch] = hn;
        cx_out[ch] = cn;
    }
}

// ---------------------------------------------------------------------------
// Host-side orchestration (graph-capture safe: only async launches on stream).
// ---------------------------------------------------------------------------
extern "C" void kernel_launch(void* const* d_in, const int* in_sizes, int n_in,
                              void* d_out, int out_size, void* d_ws, size_t ws_size,
                              hipStream_t stream)
{
    (void)in_sizes; (void)n_in; (void)out_size; (void)ws_size;

    const float* X  = (const float*)d_in[0];
    const float* Wf = (const float*)d_in[1];
    const float* bf_ = (const float*)d_in[2];
    const float* rf = (const float*)d_in[3];
    const float* Wi = (const float*)d_in[4];
    const float* bi_ = (const float*)d_in[5];
    const float* ri = (const float*)d_in[6];
    const float* Wu = (const float*)d_in[7];
    const float* bu_ = (const float*)d_in[8];
    const float* ru = (const float*)d_in[9];
    const float* Wo = (const float*)d_in[10];
    const float* bo_ = (const float*)d_in[11];
    const float* ro = (const float*)d_in[12];
    float* out = (float*)d_out;

    // Workspace layout (all offsets 64B aligned by construction).
    char* p = (char*)d_ws;
    bf16*  Wall = (bf16*)p;   p += (size_t)NGATE * CDIM * sizeof(bf16);        // 1.5 MB
    float* bias = (float*)p;  p += (size_t)NGATE * sizeof(float);              // 4 KB
    bf16*  hb   = (bf16*)p;   p += (size_t)BATCH * HID * sizeof(bf16);         // 128 KB
    float* cbuf = (float*)p;  p += (size_t)BATCH * HID * sizeof(float);        // 256 KB
    float* pre  = (float*)p;  p += (size_t)BATCH * NGATE * sizeof(float);      // 1 MB
    bf16*  Xb   = (bf16*)p;   p += (size_t)SEQ * BATCH * IN_DIM * sizeof(bf16);// 128 MB
    float* P    = (float*)p;  /* SEQ*BATCH*NGATE f32 = 512 MB */

    // (0) pack weights/bias, zero state
    pack_kernel<<<(NGATE * CDIM + 255) / 256, 256, 0, stream>>>(
        Wf, bf_, rf, Wi, bi_, ri, Wu, bu_, ru, Wo, bo_, ro, Wall, bias, hb, cbuf);

    // (1) X -> bf16
    cvt_x_kernel<<<(SEQ * BATCH * IN_DIM / 4 + 255) / 256, 256, 0, stream>>>(X, Xb);

    // (2) parallel input projection for ALL timesteps:
    //     P[t*B+b, n] = sum_{k<512} Xb[t,b,k] * Wall[n,k]
    gemm_wmma_kernel<<<dim3(NGATE / BN, (SEQ * BATCH) / BM), 256, 0, stream>>>(
        Xb, IN_DIM, Wall, CDIM, P, NGATE, nullptr, nullptr, IN_DIM);

    // (3) sequential recurrence: pre = h @ Wh^T + P[t] + bias ; cell update
    float* hx_out = out + (size_t)SEQ * BATCH * HID;
    float* cx_out = hx_out + (size_t)BATCH * HID;
    for (int t = 0; t < SEQ; ++t) {
        gemm_wmma_kernel<<<dim3(NGATE / BN, BATCH / BM), 256, 0, stream>>>(
            hb, HID, Wall + IN_DIM /* k-offset 512 into rows */, CDIM, pre, NGATE,
            P + (size_t)t * BATCH * NGATE, bias, HID);
        step_update_kernel<<<BATCH, HID, 0, stream>>>(
            pre, cbuf, hb, out + (size_t)t * BATCH * HID, hx_out, cx_out, t == SEQ - 1);
    }
}
